// Attention_40630390620348
// MI455X (gfx1250) — matrix operand order repaired and checked by reading, not compiled, not run on hardware
//
#include <hip/hip_runtime.h>
#include <hip/hip_bf16.h>

typedef __attribute__((ext_vector_type(16))) __bf16 v16bf;
typedef __attribute__((ext_vector_type(8)))  __bf16 v8bf;
typedef __attribute__((ext_vector_type(4)))  __bf16 v4bf;
typedef __attribute__((ext_vector_type(8)))  float  v8f;

#define NB 2
#define NH 16
#define SEQ 2048
#define HD 128

#define BLOCK_M 128
#define BLOCK_N 32
#define NWAVES 8

// LDS row strides in __bf16 elements (16B-aligned rows + conflict-free banks)
#define K_STRIDE 136   // 272 B/row, 32 rows  (K block, row-major)
#define VT_STRIDE 40   // 80 B/row, 128 rows  (V block, transposed)
#define P_STRIDE 40    // 80 B/row, 16 rows per wave (probabilities)

__device__ __forceinline__ float half_reduce_max(float x) {
  x = fmaxf(x, __shfl_xor(x, 1, 32));
  x = fmaxf(x, __shfl_xor(x, 2, 32));
  x = fmaxf(x, __shfl_xor(x, 4, 32));
  x = fmaxf(x, __shfl_xor(x, 8, 32));
  return x;
}
__device__ __forceinline__ float half_reduce_sum(float x) {
  x += __shfl_xor(x, 1, 32);
  x += __shfl_xor(x, 2, 32);
  x += __shfl_xor(x, 4, 32);
  x += __shfl_xor(x, 8, 32);
  return x;
}

__global__ __launch_bounds__(256, 1)
void fa_fwd_kernel(const float* __restrict__ Q, const float* __restrict__ K,
                   const float* __restrict__ V, float* __restrict__ O)
{
  __shared__ __bf16 sK[BLOCK_N * K_STRIDE];       // 8704 B
  __shared__ __bf16 sVT[HD * VT_STRIDE];          // 10240 B
  __shared__ __bf16 sP[NWAVES * 16 * P_STRIDE];   // 10240 B

  const int tid  = threadIdx.x;
  const int wave = tid >> 5;
  const int lane = tid & 31;
  const int hl   = lane >> 4;    // wave half (0: lanes 0-15, 1: lanes 16-31)
  const int l16  = lane & 15;

  const int bh   = blockIdx.y;   // b*NH + h
  const int mblk = blockIdx.x;
  const size_t base = (size_t)bh * SEQ * HD;
  const float* Qb = Q + base;
  const float* Kb = K + base;
  const float* Vb = V + base;

  const int q0w = mblk * BLOCK_M + wave * 16;  // wave's first query row

  // net score scale: QK/(sqrt(d)*coeff)*coeff = QK/sqrt(128). Folded into Q here
  // (linear in Q), so the inner loop does no per-element scaling.
  const float scale = 0.088388347648318447f;

  // ---- Q -> register-resident bf16 A-fragments (4 K-chunks of 32 over D=128)
  // 16-bit A layout: elems 0..15 = K = 16*hl + 0..15 (contiguous, matches B)
  v16bf qa[4];
  {
    const float* qrow = Qb + (size_t)(q0w + l16) * HD;
#pragma unroll
    for (int kc = 0; kc < 4; ++kc) {
      const float* qp = qrow + kc * 32 + hl * 16;
#pragma unroll
      for (int j = 0; j < 8; ++j) {
        qa[kc][j]     = (__bf16)(qp[j] * scale);
        qa[kc][8 + j] = (__bf16)(qp[8 + j] * scale);
      }
    }
  }

  v8f acc[8];
#pragma unroll
  for (int n = 0; n < 8; ++n)
#pragma unroll
    for (int j = 0; j < 8; ++j) acc[n][j] = 0.0f;

  float mrow[8], lrow[8];
#pragma unroll
  for (int v = 0; v < 8; ++v) { mrow[v] = -__builtin_inff(); lrow[v] = 0.0f; }

  const int NKB = mblk * 4 + 4;            // key blocks needed by whole workgroup
  const int myLimit = (q0w + 15) >> 5;     // last key block this wave touches

  for (int kb = 0; kb < NKB; ++kb) {
    __syncthreads();  // previous iteration's LDS reads complete before restaging
    // ---- cooperative stage: K row-major bf16, V transposed bf16 ----
#pragma unroll
    for (int it = 0; it < 4; ++it) {
      const int f    = tid + it * 256;     // float4 id in [0, 1024)
      const int krow = f >> 5;             // 0..31
      const int c4   = f & 31;             // float4 column
      const float* kg = Kb + (size_t)(kb * BLOCK_N + krow) * HD;
      const float* vg = Vb + (size_t)(kb * BLOCK_N + krow) * HD;
      const float4 k4 = ((const float4*)kg)[c4];
      v4bf kq; kq[0]=(__bf16)k4.x; kq[1]=(__bf16)k4.y; kq[2]=(__bf16)k4.z; kq[3]=(__bf16)k4.w;
      *(v4bf*)&sK[krow * K_STRIDE + c4 * 4] = kq;
      const float4 vv = ((const float4*)vg)[c4];
      sVT[(c4*4+0) * VT_STRIDE + krow] = (__bf16)vv.x;
      sVT[(c4*4+1) * VT_STRIDE + krow] = (__bf16)vv.y;
      sVT[(c4*4+2) * VT_STRIDE + krow] = (__bf16)vv.z;
      sVT[(c4*4+3) * VT_STRIDE + krow] = (__bf16)vv.w;
      // fire-and-forget prefetch of the next key block (global_prefetch_b8)
      if (kb + 1 < NKB) {
        __builtin_prefetch(kg + (size_t)BLOCK_N * HD + c4 * 4, 0, 0);
        __builtin_prefetch(vg + (size_t)BLOCK_N * HD + c4 * 4, 0, 0);
      }
    }
    __syncthreads();

    const bool active = (kb <= myLimit);   // wave-uniform: EXEC stays all-1s
    if (active) {
      // ---- S = Q K^T  (16 q-rows x 32 keys = two 16x16 C tiles) ----
      v8f s0, s1;
#pragma unroll
      for (int j = 0; j < 8; ++j) { s0[j] = 0.0f; s1[j] = 0.0f; }
#pragma unroll
      for (int kc = 0; kc < 4; ++kc) {
        // B fragment: lane N = key, elems 0..15 = contiguous d = kc*32 + 16*hl + 0..15
        const __bf16* kp0 = &sK[l16 * K_STRIDE + kc * 32 + hl * 16];
        const __bf16* kp1 = kp0 + 16 * K_STRIDE;
        v8bf b0l = *(const v8bf*)kp0, b0h = *(const v8bf*)(kp0 + 8);
        v8bf b1l = *(const v8bf*)kp1, b1h = *(const v8bf*)(kp1 + 8);
        v16bf b0 = __builtin_shufflevector(b0l, b0h, 0,1,2,3,4,5,6,7,8,9,10,11,12,13,14,15);
        v16bf b1 = __builtin_shufflevector(b1l, b1h, 0,1,2,3,4,5,6,7,8,9,10,11,12,13,14,15);
        s0 = __builtin_amdgcn_wmma_f32_16x16x32_bf16(false, qa[kc], false, b0, (short)0, s0, false, false);
        s1 = __builtin_amdgcn_wmma_f32_16x16x32_bf16(false, qa[kc], false, b1, (short)0, s1, false, false);
      }
      // ---- causal mask: only the diagonal block, behind a wave-uniform branch ----
      if (kb == myLimit) {
#pragma unroll
        for (int v = 0; v < 8; ++v) {
          const int qg = q0w + v + 8 * hl;   // C-layout row for this element
          if (kb * BLOCK_N + l16      > qg) s0[v] = -__builtin_inff();
          if (kb * BLOCK_N + 16 + l16 > qg) s1[v] = -__builtin_inff();
        }
      }
      // ---- online softmax over this 32-key block (fp32 stats) ----
      const int prow = wave * 16;
#pragma unroll
      for (int v = 0; v < 8; ++v) {
        const float x0 = s0[v];
        const float x1 = s1[v];
        const float mx = half_reduce_max(fmaxf(x0, x1));
        const float mn = fmaxf(mrow[v], mx);
        const float p0 = __expf(x0 - mn);
        const float p1 = __expf(x1 - mn);
        const float c  = __expf(mrow[v] - mn);
        lrow[v] = lrow[v] * c + half_reduce_sum(p0 + p1);
        mrow[v] = mn;
        const int r = prow + v + 8 * hl;
        sP[r * P_STRIDE + l16]      = (__bf16)p0;
        sP[r * P_STRIDE + 16 + l16] = (__bf16)p1;
#pragma unroll
        for (int n = 0; n < 8; ++n) acc[n][v] *= c;
      }
    }
    __syncthreads();  // P visible as A-fragments; also orders sVT reads below
    if (active) {
      // ---- O += P V : A = P (16x32), B = V^T rows (contiguous keys) ----
      const __bf16* pp = &sP[(wave * 16 + l16) * P_STRIDE + hl * 16];
      v8bf pl = *(const v8bf*)pp;
      v8bf ph = *(const v8bf*)(pp + 8);
      v16bf pa = __builtin_shufflevector(pl, ph, 0,1,2,3,4,5,6,7,8,9,10,11,12,13,14,15);
#pragma unroll
      for (int n = 0; n < 8; ++n) {
        const __bf16* vp = &sVT[(n * 16 + l16) * VT_STRIDE + hl * 16];
        v8bf vl = *(const v8bf*)vp, vh = *(const v8bf*)(vp + 8);
        v16bf bv = __builtin_shufflevector(vl, vh, 0,1,2,3,4,5,6,7,8,9,10,11,12,13,14,15);
        acc[n] = __builtin_amdgcn_wmma_f32_16x16x32_bf16(false, pa, false, bv, (short)0, acc[n], false, false);
      }
    }
  }

  // ---- epilogue: normalize, write out[b, q, h*128 + d] ----
  const int b = bh / NH;
  const int h = bh % NH;
  float inv[8];
#pragma unroll
  for (int v = 0; v < 8; ++v) inv[v] = 1.0f / lrow[v];
#pragma unroll
  for (int n = 0; n < 8; ++n) {
#pragma unroll
    for (int v = 0; v < 8; ++v) {
      const int q = q0w + v + 8 * hl;
      const int d = n * 16 + l16;
      O[((size_t)b * SEQ + q) * (NH * HD) + h * HD + d] = acc[n][v] * inv[v];
    }
  }
}

extern "C" void kernel_launch(void* const* d_in, const int* in_sizes, int n_in,
                              void* d_out, int out_size, void* d_ws, size_t ws_size,
                              hipStream_t stream)
{
  const float* Q = (const float*)d_in[0];
  const float* K = (const float*)d_in[1];
  const float* V = (const float*)d_in[2];
  float* O = (float*)d_out;
  dim3 grid(SEQ / BLOCK_M, NB * NH);
  dim3 block(256);
  fa_fwd_kernel<<<grid, block, 0, stream>>>(Q, K, V, O);
}
